// SimpleLanguageModel_31258771980353
// MI455X (gfx1250) — matrix operand order repaired and checked by reading, not compiled, not run on hardware
//
#include <hip/hip_runtime.h>

typedef __attribute__((ext_vector_type(16))) _Float16 v16h;
typedef __attribute__((ext_vector_type(8)))  _Float16 v8h;
typedef __attribute__((ext_vector_type(8)))  float    v8f;

#define BATCH   131072
#define FDIM    512
#define F2      256
#define NOBJ    100000
#define NPRED   2000
#define EPSV    1e-5f
#define ROWS_PB 32

// ---- workspace layout (bytes) ----
#define OFF_Z      0u                         // BATCH*256*4 = 134,217,728
#define OFF_COUNTS 134217728u                 // (100000+100000+2000)*4
#define OFF_SUMS   135025920u                 // 3072 f32 (sum/ss for 3 streams)
#define OFF_COEF   135038208u                 // 3072 f32 (a/c for 3 streams)
#define OFF_ZSTAT  135050496u                 // 512 f32
#define OFF_ZCOEF  135052544u                 // 512 f32
#define OFF_BP     135054592u                 // 131072 halves (W4 packed, 32B aligned)

__global__ void k_zero(int* __restrict__ p, int n) {
    int i = blockIdx.x * 256 + threadIdx.x;
    if (i < n) p[i] = 0;
}

__global__ void k_hist(const int* __restrict__ subj, const int* __restrict__ obj,
                       const int* __restrict__ predi,
                       int* __restrict__ cS, int* __restrict__ cO, int* __restrict__ cP) {
    int i = blockIdx.x * 256 + threadIdx.x;
    atomicAdd(&cS[subj[i]], 1);
    atomicAdd(&cO[obj[i]], 1);
    atomicAdd(&cP[predi[i]], 1);
}

// count-weighted per-feature sum & sum-of-squares of one weight table
__global__ void k_scan(const float* __restrict__ W, const int* __restrict__ counts,
                       int rows, float* __restrict__ sums) {
    int tid = threadIdx.x;
    int f = tid * 2;
    int r0 = blockIdx.x * 64;
    float s0 = 0.f, s1 = 0.f, q0 = 0.f, q1 = 0.f;
    for (int r = 0; r < 64; ++r) {
        int row = r0 + r;
        if (row >= rows) break;
        float c = (float)counts[row];
        if (c == 0.f) continue;
        float2 w = *(const float2*)&W[(size_t)row * FDIM + f];
        s0 += c * w.x;        s1 += c * w.y;
        q0 += c * w.x * w.x;  q1 += c * w.y * w.y;
    }
    atomicAdd(&sums[f],       s0);
    atomicAdd(&sums[f + 1],   s1);
    atomicAdd(&sums[512 + f],     q0);
    atomicAdd(&sums[512 + f + 1], q1);
}

// BN(x)=a*x+c with bias terms cancelled by mean subtraction
__global__ void k_coef(const float* __restrict__ sums,
                       const float* __restrict__ g1, const float* __restrict__ be1,
                       const float* __restrict__ g2, const float* __restrict__ be2,
                       const float* __restrict__ g3, const float* __restrict__ be3,
                       float* __restrict__ coef) {
    int i = blockIdx.x * 256 + threadIdx.x;
    if (i >= 1536) return;
    int s = i >> 9, f = i & 511;
    float mean = sums[s * 1024 + f] * (1.f / BATCH);
    float var  = sums[s * 1024 + 512 + f] * (1.f / BATCH) - mean * mean;
    const float* g  = (s == 0) ? g1  : (s == 1) ? g2  : g3;
    const float* be = (s == 0) ? be1 : (s == 1) ? be2 : be3;
    float a = g[f] * rsqrtf(var + EPSV);
    coef[s * 1024 + f]       = a;
    coef[s * 1024 + 512 + f] = be[f] - a * mean;
}

// pack W4 (256x512 f32, row=out) into WMMA 16-bit B fragments:
// fragment (ntile,kstep): lane l holds N = ntile*16+(l&15), K = kstep*32+(l>>4)*16+h
__global__ void k_pack(const float* __restrict__ W4, _Float16* __restrict__ Bp) {
    int idx = blockIdx.x * 256 + threadIdx.x;      // 8192 lane-entries
    int frag = idx >> 5, lane = idx & 31;
    int ntile = frag >> 4, kstep = frag & 15;
    int N  = ntile * 16 + (lane & 15);
    int K0 = kstep * 32 + ((lane >> 4) << 4);
    _Float16* dst = Bp + (size_t)idx * 16;
#pragma unroll
    for (int h = 0; h < 16; ++h)
        dst[h] = (_Float16)W4[(size_t)N * FDIM + K0 + h];
}

__global__ void k_main(const int* __restrict__ subj, const int* __restrict__ obj,
                       const int* __restrict__ predi,
                       const float* __restrict__ W1, const float* __restrict__ W2,
                       const float* __restrict__ W3,
                       const float* __restrict__ coef, const float* __restrict__ b4,
                       const v16h* __restrict__ Bp,
                       float* __restrict__ z, float* __restrict__ zstat) {
    __shared__ __attribute__((aligned(32))) _Float16 Atile[ROWS_PB * FDIM]; // 32 KB
    __shared__ float cf[3072];                                              // 12 KB
    __shared__ float zp[512];                                               //  2 KB

    int tid = threadIdx.x;
    for (int i = tid; i < 3072; i += 256) cf[i] = coef[i];
    zp[tid] = 0.f; zp[256 + tid] = 0.f;
    __syncthreads();

    int row0 = blockIdx.x * ROWS_PB;

    // ---- gather + BN-affine + ReLU + sum -> f16 A-fragments in LDS ----
    {
        int rl = tid >> 3, q = tid & 7;            // 32 rows x 8 threads/row
        int row = row0 + rl;
        size_t is = (size_t)subj[row] * FDIM;
        size_t io = (size_t)obj[row]  * FDIM;
        size_t ip = (size_t)predi[row] * FDIM;
        int mtile = rl >> 4, m = rl & 15;
        for (int g = 0; g < 8; ++g) {
            int k0 = q * 64 + g * 8;
            v8h hv;
#pragma unroll
            for (int e = 0; e < 8; ++e) {
                int k = k0 + e;
                float f1 = fmaxf(cf[k]        * W1[is + k] + cf[512 + k],  0.f);
                float f2 = fmaxf(cf[1024 + k] * W2[io + k] + cf[1536 + k], 0.f);
                float f3 = fmaxf(cf[2048 + k] * W3[ip + k] + cf[2560 + k], 0.f);
                hv[e] = (_Float16)(f1 + f2 + f3);
            }
            // A-fragment address: lane = m + 16*(grp>>1), halves hb..hb+7
            int kstep = k0 >> 5, kk = k0 & 31, grp = kk >> 3;
            int lane = m + ((grp >> 1) << 4);
            int hb   = (grp & 1) << 3;
            *(v8h*)&Atile[(mtile * 16 + kstep) * 512 + lane * 16 + hb] = hv;
        }
    }
    __syncthreads();

    // ---- WMMA: 8 waves x (2 mtiles x 2 ntiles) tiles, K = 16 steps of 32 ----
    {
        int wave = tid >> 5, lane = tid & 31;
        for (int t = 0; t < 4; ++t) {
            int nt = wave * 2 + (t & 1);
            int mt = t >> 1;
            v8f acc = {};
            for (int ks = 0; ks < 16; ++ks) {
                v16h a = *(const v16h*)&Atile[(mt * 16 + ks) * 512 + lane * 16];
                v16h b = Bp[(nt * 16 + ks) * 32 + lane];
                acc = __builtin_amdgcn_wmma_f32_16x16x32_f16(
                        false, a, false, b, (short)0, acc, false, false);
            }
            int col = nt * 16 + (lane & 15);
            float bias = b4[col];
            int rbase = row0 + mt * 16 + ((lane >= 16) ? 8 : 0);
            float ls = 0.f, lq = 0.f;
#pragma unroll
            for (int j = 0; j < 8; ++j) {
                float zv = acc[j] + bias;
                z[(size_t)(rbase + j) * F2 + col] = zv;
                ls += zv; lq += zv * zv;
            }
            atomicAdd(&zp[col],       ls);
            atomicAdd(&zp[256 + col], lq);
        }
    }
    __syncthreads();
    atomicAdd(&zstat[tid],       zp[tid]);
    atomicAdd(&zstat[256 + tid], zp[256 + tid]);
}

__global__ void k_zfin(const float* __restrict__ zstat,
                       const float* __restrict__ g4, const float* __restrict__ be4,
                       float* __restrict__ zcoef) {
    int f = threadIdx.x;                   // 256
    float mean = zstat[f] * (1.f / BATCH);
    float var  = zstat[256 + f] * (1.f / BATCH) - mean * mean;
    float a = g4[f] * rsqrtf(var + EPSV);
    zcoef[f]       = a;
    zcoef[256 + f] = be4[f] - a * mean;
}

__global__ void k_logit(const float* __restrict__ z, const float* __restrict__ zcoef,
                        const float* __restrict__ W5, const float* __restrict__ b5,
                        float* __restrict__ out) {
    __shared__ float la[256], lc[256], lw[256];
    int tid = threadIdx.x;
    la[tid] = zcoef[tid];
    lc[tid] = zcoef[256 + tid];
    lw[tid] = W5[tid];
    __syncthreads();
    size_t row = (size_t)blockIdx.x * 256 + tid;
    const float* zr = z + row * F2;
    float acc = 0.f;
    for (int k = 0; k < 256; k += 4) {
        float4 v = *(const float4*)&zr[k];
        acc += fmaxf(la[k]     * v.x + lc[k],     0.f) * lw[k];
        acc += fmaxf(la[k + 1] * v.y + lc[k + 1], 0.f) * lw[k + 1];
        acc += fmaxf(la[k + 2] * v.z + lc[k + 2], 0.f) * lw[k + 2];
        acc += fmaxf(la[k + 3] * v.w + lc[k + 3], 0.f) * lw[k + 3];
    }
    out[row] = acc + b5[0];
}

extern "C" void kernel_launch(void* const* d_in, const int* in_sizes, int n_in,
                              void* d_out, int out_size, void* d_ws, size_t ws_size,
                              hipStream_t stream) {
    const int*   subj  = (const int*)d_in[0];
    const int*   obj   = (const int*)d_in[1];
    const int*   predi = (const int*)d_in[2];
    const float* W1 = (const float*)d_in[3];   // b1 (d_in[4]) cancels in BN
    const float* g1 = (const float*)d_in[5];
    const float* be1= (const float*)d_in[6];
    const float* W2 = (const float*)d_in[7];   // b2 cancels
    const float* g2 = (const float*)d_in[9];
    const float* be2= (const float*)d_in[10];
    const float* W3 = (const float*)d_in[11];  // b3 cancels
    const float* g3 = (const float*)d_in[13];
    const float* be3= (const float*)d_in[14];
    const float* W4 = (const float*)d_in[15];
    const float* b4 = (const float*)d_in[16];  // cancels too, kept for fidelity
    const float* g4 = (const float*)d_in[17];
    const float* be4= (const float*)d_in[18];
    const float* W5 = (const float*)d_in[19];
    const float* b5 = (const float*)d_in[20];
    float* out = (float*)d_out;

    char* ws = (char*)d_ws;
    float*    z     = (float*)(ws + OFF_Z);
    int*      cS    = (int*)(ws + OFF_COUNTS);
    int*      cO    = cS + NOBJ;
    int*      cP    = cO + NOBJ;
    float*    sums  = (float*)(ws + OFF_SUMS);
    float*    coef  = (float*)(ws + OFF_COEF);
    float*    zstat = (float*)(ws + OFF_ZSTAT);
    float*    zcoef = (float*)(ws + OFF_ZCOEF);
    _Float16* Bp    = (_Float16*)(ws + OFF_BP);

    int zero_words = (int)((OFF_BP - OFF_COUNTS) / 4);
    k_zero<<<(zero_words + 255) / 256, 256, 0, stream>>>((int*)(ws + OFF_COUNTS), zero_words);
    k_hist<<<BATCH / 256, 256, 0, stream>>>(subj, obj, predi, cS, cO, cP);
    k_scan<<<(NOBJ + 63) / 64, 256, 0, stream>>>(W1, cS, NOBJ, sums);
    k_scan<<<(NOBJ + 63) / 64, 256, 0, stream>>>(W2, cO, NOBJ, sums + 1024);
    k_scan<<<(NPRED + 63) / 64, 256, 0, stream>>>(W3, cP, NPRED, sums + 2048);
    k_coef<<<6, 256, 0, stream>>>(sums, g1, be1, g2, be2, g3, be3, coef);
    k_pack<<<32, 256, 0, stream>>>(W4, Bp);
    k_main<<<BATCH / ROWS_PB, 256, 0, stream>>>(subj, obj, predi, W1, W2, W3,
                                                coef, b4, (const v16h*)Bp, z, zstat);
    k_zfin<<<1, 256, 0, stream>>>(zstat, g4, be4, zcoef);
    k_logit<<<BATCH / 256, 256, 0, stream>>>(z, zcoef, W5, b5, out);
}
